// IAF_61684320305496
// MI455X (gfx1250) — matrix operand order repaired and checked by reading, not compiled, not run on hardware
//
#include <hip/hip_runtime.h>

// ---------------------------------------------------------------------------
// IAF (B=512, D=256, 4 flows) for MI455X / gfx1250.
//
// Dominant cost: 16 batched 256x256x256 GEMMs over 512 batches (~275 GFLOP).
// Strategy: Jacobian chain in bf16 (fp32 accumulate) via
// V_WMMA_F32_16X16X32_BF16 (wave32). Jac buffer = 64MB bf16 -> each batched
// GEMM moves ~128MB; 16 GEMMs ~ 2GB / 23.3TB/s ~ 90us, memory bound, with
// the ~200MB working set largely resident in the 192MB L2.
//
// Round 3: the 64Nx256K B-strip is staged once per block into LDS (32KB,
// padded rows -> conflict-free ds_load_b128 fragment reads shared by all 8
// waves); only the per-wave A stream reads global memory, double-buffered in
// registers. This removes the per-K-step global-load drains seen in round 2.
// ---------------------------------------------------------------------------

#define DD   256
#define BB   512
#define NF   4
#define LDSPITCH 264   // 256 + 8 bf16 pad: row stride 528B -> 64-bank spread

typedef __attribute__((ext_vector_type(16))) __bf16 v16bf;
typedef __attribute__((ext_vector_type(8)))  float  v8f;

union FragBF {
    v16bf v;
    uint4 q[2];
};

enum { MODE_COLSCALE = 0, MODE_ZK = 1, MODE_PLAIN = 2 };

__device__ __forceinline__ unsigned short f2bf(float f) {
    unsigned u = __float_as_uint(f);
    unsigned r = u + 0x7FFFu + ((u >> 16) & 1u);   // round-to-nearest-even
    return (unsigned short)(r >> 16);
}
__device__ __forceinline__ float bf2f(unsigned short h) {
    return __uint_as_float(((unsigned)h) << 16);
}

// ---------------------------------------------------------------------------
// Batched C[b] = A[b] (256x256 bf16, row-major) @ B (256x256), fp32 accum.
// Bt holds B^T row-major; bStrideBt = 0 for a shared weight, D*D for a
// per-batch right operand. Wave: 32(M) x 64(N); block: 8 waves -> 256x64.
// Grid: (256/64, 1, BB).
// ---------------------------------------------------------------------------
__global__ __launch_bounds__(256)
void bgemm_wmma(const unsigned short* __restrict__ A,
                const unsigned short* __restrict__ Bt,
                unsigned short* __restrict__ Out,
                long bStrideBt, int mode,
                const float* __restrict__ s0,   // colscale g   | c1
                const float* __restrict__ s1,   //              | cdiag
                const float* __restrict__ s2)   //              | gate
{
    __shared__ unsigned short bsm[64 * LDSPITCH];   // 64 Bt rows x 256 K (+pad)

    const int b    = blockIdx.z;
    const int wave = threadIdx.x >> 5;
    const int lane = threadIdx.x & 31;
    const int half = lane >> 4;
    const int l16  = lane & 15;
    const int m0    = wave * 32;
    const int nBase = blockIdx.x * 64;

    const unsigned short* Ab  = A  + (size_t)b * DD * DD;
    const unsigned short* Btb = Bt + (size_t)b * bStrideBt;

    // ---- Cooperative stage of the 64x256 Bt strip into LDS (coalesced). ----
    {
        const int row = threadIdx.x >> 2;            // 0..63
        const int seg = (threadIdx.x & 3) * 64;      // 4 x 64 bf16 per row
        const unsigned short* src = Btb + (size_t)(nBase + row) * DD + seg;
        unsigned short* dst = &bsm[row * LDSPITCH + seg];
#pragma unroll
        for (int q = 0; q < 8; ++q)
            *reinterpret_cast<uint4*>(dst + q * 8) =
                *reinterpret_cast<const uint4*>(src + q * 8);
    }
    __syncthreads();

    const unsigned short* ap0 = Ab + (size_t)(m0 + l16) * DD + half * 8;
    const unsigned short* ap1 = ap0 + (size_t)16 * DD;

    v8f acc[8] = {};

    // ---- K loop: A double-buffered from global, B fragments from LDS. ----
    FragBF aC0, aC1, aN0, aN1;
    aC0.q[0] = *reinterpret_cast<const uint4*>(ap0);
    aC0.q[1] = *reinterpret_cast<const uint4*>(ap0 + 16);
    aC1.q[0] = *reinterpret_cast<const uint4*>(ap1);
    aC1.q[1] = *reinterpret_cast<const uint4*>(ap1 + 16);

#pragma unroll
    for (int k0 = 0; k0 < DD; k0 += 32) {
        const bool last = (k0 + 32 >= DD);
        if (!last) {
            aN0.q[0] = *reinterpret_cast<const uint4*>(ap0 + k0 + 32);
            aN0.q[1] = *reinterpret_cast<const uint4*>(ap0 + k0 + 48);
            aN1.q[0] = *reinterpret_cast<const uint4*>(ap1 + k0 + 32);
            aN1.q[1] = *reinterpret_cast<const uint4*>(ap1 + k0 + 48);
        }
#pragma unroll
        for (int t = 0; t < 4; ++t) {
            // B fragment (32x16): lane fixed N, K = half*8+0..7 and 16+half*8+0..7 from LDS row.
            FragBF bf;
            const unsigned short* bp =
                &bsm[(t * 16 + l16) * LDSPITCH + k0 + half * 8];
            bf.q[0] = *reinterpret_cast<const uint4*>(bp);
            bf.q[1] = *reinterpret_cast<const uint4*>(bp + 16);
            acc[t] = __builtin_amdgcn_wmma_f32_16x16x32_bf16(
                false, aC0.v, false, bf.v, (short)0, acc[t], false, false);
            acc[4 + t] = __builtin_amdgcn_wmma_f32_16x16x32_bf16(
                false, aC1.v, false, bf.v, (short)0, acc[4 + t], false, false);
        }
        if (!last) { aC0 = aN0; aC1 = aN1; }
    }

    // Epilogue. Accumulator acc[mh*4+t]: lane has fixed N = nBase+t*16+l16,
    // rows M = m0 + mh*16 + half*8 + r (r = 0..7).
#pragma unroll
    for (int mh = 0; mh < 2; ++mh) {
#pragma unroll
        for (int t = 0; t < 4; ++t) {
            const v8f& av = acc[mh * 4 + t];
            const int n = nBase + t * 16 + l16;
            const int mrow = m0 + mh * 16 + half * 8;
            if (mode == MODE_ZK) {
                // jac_zkp1(m,n) = acc*c1[n] + (m==n)*(acc*cdiag[n] + gate[n]),
                // stored TRANSPOSED (Zt[n][m]) -> contiguous 16B store/lane,
                // exactly the B^T layout the next GEMM consumes.
                const float c1 = s0[(size_t)b * DD + n];
                const float cd = s1[(size_t)b * DD + n];
                const float gt = s2[(size_t)b * DD + n];
                unsigned short tmp[8];
#pragma unroll
                for (int r = 0; r < 8; ++r) {
                    const int m = mrow + r;
                    float v = av[r] * c1;
                    if (m == n) v += av[r] * cd + gt;
                    tmp[r] = f2bf(v);
                }
                *reinterpret_cast<uint4*>(Out + (size_t)b * DD * DD +
                                          (size_t)n * DD + mrow) =
                    *reinterpret_cast<const uint4*>(tmp);
            } else {
                const float g = (mode == MODE_COLSCALE) ? s0[(size_t)b * DD + n] : 1.0f;
                unsigned short* o = Out + (size_t)b * DD * DD + n;
#pragma unroll
                for (int r = 0; r < 8; ++r) {
                    o[(size_t)(mrow + r) * DD] = f2bf(av[r] * g);  // lanes coalesce per row
                }
            }
        }
    }
}

// ---------------------------------------------------------------------------
// Pre-mask + transpose weights to bf16: Wt[f][j][n][k] = mask * W[f][k][n].
// j = 0: triu(W0), 1: triu(W1), 2: strict-triu(Wm).
// ---------------------------------------------------------------------------
__global__ __launch_bounds__(256)
void prep_weights(const float* __restrict__ W0, const float* __restrict__ W1,
                  const float* __restrict__ Wm, unsigned short* __restrict__ Wt)
{
    const int k = threadIdx.x;
    const int n = blockIdx.x;
    const int f = blockIdx.y;
    const size_t win  = (size_t)f * DD * DD + (size_t)k * DD + n;
    const size_t wout = (size_t)f * 3 * DD * DD + (size_t)n * DD + k;
    Wt[wout]               = (k <= n) ? f2bf(W0[win]) : (unsigned short)0;
    Wt[wout + DD * DD]     = (k <= n) ? f2bf(W1[win]) : (unsigned short)0;
    Wt[wout + 2 * DD * DD] = (k <  n) ? f2bf(Wm[win]) : (unsigned short)0;
}

// ---------------------------------------------------------------------------
// Small triangular GEMM + activation:  Y = act(X (+Xadd) @ mask*W + bias)
// act: 0 = elu (also writes elu' to G), 1 = linear, 2 = sigmoid(.+1.0)
// ---------------------------------------------------------------------------
__global__ __launch_bounds__(256)
void small_gemm(const float* __restrict__ X, const float* __restrict__ Xadd,
                const float* __restrict__ W, const float* __restrict__ bias,
                float* __restrict__ Y, float* __restrict__ G,
                int strict, int act)
{
    __shared__ float xs[DD];
    const int b = blockIdx.x;
    const int n = threadIdx.x;
    float xv = X[(size_t)b * DD + n];
    if (Xadd) xv += Xadd[(size_t)b * DD + n];
    xs[n] = xv;
    __syncthreads();

    float accv = bias[n];
    const int kend = strict ? n : n + 1;
    for (int k = 0; k < kend; ++k)
        accv = fmaf(xs[k], W[(size_t)k * DD + n], accv);

    const size_t i = (size_t)b * DD + n;
    if (act == 0) {
        const float h = accv > 0.0f ? accv : expm1f(accv);
        Y[i] = h;
        G[i] = (h > 0.0f ? 1.0f : 0.0f) + (h < 0.0f ? h + 1.0f : 0.0f);
    } else if (act == 1) {
        Y[i] = accv;
    } else {
        Y[i] = 1.0f / (1.0f + __expf(-(accv + 1.0f)));   // FORGET_BIAS = 1.0
    }
}

// z update, epilogue coefficients, logdet accumulation.
__global__ __launch_bounds__(256)
void pointwise(float* __restrict__ z, const float* __restrict__ mean,
               const float* __restrict__ gate,
               float* __restrict__ c1, float* __restrict__ cdiag,
               float* __restrict__ logdets)
{
    __shared__ float red[DD];
    const int b = blockIdx.x;
    const int d = threadIdx.x;
    const size_t i = (size_t)b * DD + d;
    const float zc = z[i], mn = mean[i], gt = gate[i];
    const float zn = gt * zc + (1.0f - gt) * mn;
    const float sg = gt * (1.0f - gt);
    z[i]     = zn;
    c1[i]    = -sg * mn;
    cdiag[i] = sg * zn + 1.0f - gt;
    red[d] = __logf(gt);
    __syncthreads();
    for (int s = 128; s > 0; s >>= 1) {
        if (d < s) red[d] += red[d + s];
        __syncthreads();
    }
    if (d == 0) logdets[b] += red[0];
}

__global__ __launch_bounds__(256)
void init_state(const float* __restrict__ zin, float* __restrict__ z,
                float* __restrict__ logdets, unsigned short* __restrict__ J)
{
    const size_t idx = (size_t)blockIdx.x * blockDim.x + threadIdx.x;  // B*D*D
    const size_t m = (idx / DD) % DD, n = idx % DD;
    J[idx] = (m == n) ? (unsigned short)0x3F80 : (unsigned short)0;
    if (idx < (size_t)BB * DD) z[idx] = zin[idx];
    if (idx < BB) logdets[idx] = 0.0f;
}

__global__ void rev_z(float* __restrict__ z) {
    const int b = blockIdx.x, d = threadIdx.x;   // 128 threads
    const size_t i = (size_t)b * DD + d, j = (size_t)b * DD + (DD - 1 - d);
    const float a = z[i], c = z[j];
    z[i] = c; z[j] = a;
}

__global__ __launch_bounds__(256)
void rev_jac(unsigned short* __restrict__ J) {
    const size_t idx = (size_t)blockIdx.x * blockDim.x + threadIdx.x; // B*D*(D/2)
    const size_t n   = idx % (DD / 2);
    const size_t row = idx / (DD / 2);
    const size_t i = row * DD + n, j = row * DD + (DD - 1 - n);
    const unsigned short a = J[i], c = J[j];
    J[i] = c; J[j] = a;
}

__global__ __launch_bounds__(256)
void finalize(const float* __restrict__ z, const float* __restrict__ logdets,
              const unsigned short* __restrict__ J, float* __restrict__ out)
{
    const size_t idx = (size_t)blockIdx.x * blockDim.x + threadIdx.x;  // B*D*D
    out[(size_t)BB * DD + BB + idx] = bf2f(J[idx]);
    if (idx < (size_t)BB * DD) out[idx] = z[idx];
    if (idx < BB) out[(size_t)BB * DD + idx] = logdets[idx];
}

extern "C" void kernel_launch(void* const* d_in, const int* in_sizes, int n_in,
                              void* d_out, int out_size, void* d_ws, size_t ws_size,
                              hipStream_t stream)
{
    (void)in_sizes; (void)n_in; (void)out_size; (void)ws_size;
    const float* zin  = (const float*)d_in[0];
    const float* hctx = (const float*)d_in[1];
    const float* W0   = (const float*)d_in[2];
    const float* b0   = (const float*)d_in[3];
    const float* W1   = (const float*)d_in[4];
    const float* b1   = (const float*)d_in[5];
    const float* Wm   = (const float*)d_in[6];
    const float* bm   = (const float*)d_in[7];
    const float* Wsw  = (const float*)d_in[8];
    const float* bs   = (const float*)d_in[9];
    float* out = (float*)d_out;

    // Workspace carve-up: 3x64MB bf16 jac ping-pong + weights + fp32 vectors.
    char* ws = (char*)d_ws;
    const size_t JBYTES = (size_t)BB * DD * DD * 2;
    unsigned short* jbuf[3] = {
        (unsigned short*)ws,
        (unsigned short*)(ws + JBYTES),
        (unsigned short*)(ws + 2 * JBYTES)
    };
    char* p = ws + 3 * JBYTES;
    unsigned short* Wt = (unsigned short*)p;  p += (size_t)NF * 3 * DD * DD * 2;
    const size_t VB = (size_t)BB * DD * sizeof(float);
    float* zbuf  = (float*)p; p += VB;
    float* h1    = (float*)p; p += VB;
    float* g1    = (float*)p; p += VB;
    float* h2    = (float*)p; p += VB;
    float* g2    = (float*)p; p += VB;
    float* meanv = (float*)p; p += VB;
    float* gatev = (float*)p; p += VB;
    float* c1v   = (float*)p; p += VB;
    float* cdv   = (float*)p; p += VB;
    float* ldet  = (float*)p; p += (size_t)BB * sizeof(float);

    prep_weights<<<dim3(DD, NF), DD, 0, stream>>>(W0, W1, Wm, Wt);
    init_state<<<((size_t)BB * DD * DD) / 256, 256, 0, stream>>>(zin, zbuf, ldet, jbuf[0]);

    int Jc = 0;
    const dim3 gG(DD / 64, 1, BB), gB(256);
    for (int i = 0; i < NF; ++i) {
        if ((i + 1) % 2 == 0) {
            rev_z<<<BB, DD / 2, 0, stream>>>(zbuf);
            rev_jac<<<((size_t)BB * DD * (DD / 2)) / 256, 256, 0, stream>>>(jbuf[Jc]);
        }
        const size_t wo = (size_t)i * DD * DD, bo = (size_t)i * DD;
        small_gemm<<<BB, DD, 0, stream>>>(zbuf, nullptr, W0 + wo, b0 + bo, h1, g1, 0, 0);
        small_gemm<<<BB, DD, 0, stream>>>(h1, hctx,      W1 + wo, b1 + bo, h2, g2, 0, 0);
        small_gemm<<<BB, DD, 0, stream>>>(h2, nullptr,   Wm + wo, bm + bo, meanv, nullptr, 1, 1);
        small_gemm<<<BB, DD, 0, stream>>>(h2, nullptr,   Wsw + wo, bs + bo, gatev, nullptr, 1, 2);
        pointwise<<<BB, DD, 0, stream>>>(zbuf, meanv, gatev, c1v, cdv, ldet);

        unsigned short* WzT = Wt + ((size_t)i * 3 + 0) * DD * DD;
        unsigned short* WhT = Wt + ((size_t)i * 3 + 1) * DD * DD;
        unsigned short* WmT = Wt + ((size_t)i * 3 + 2) * DD * DD;
        const int jH = (Jc + 1) % 3, jH2 = (Jc + 2) % 3;

        // jac_h  = (jac @ triu(W0)) * elu'(h1)  (column scale in epilogue)
        bgemm_wmma<<<gG, gB, 0, stream>>>(jbuf[Jc],  WzT, jbuf[jH],  0, MODE_COLSCALE, g1, nullptr, nullptr);
        // jac_h  = (jac_h @ triu(W1)) * elu'(h2)
        bgemm_wmma<<<gG, gB, 0, stream>>>(jbuf[jH],  WhT, jbuf[jH2], 0, MODE_COLSCALE, g2, nullptr, nullptr);
        // jac_zkp1 (fused epilogue), stored transposed into jbuf[jH]
        bgemm_wmma<<<gG, gB, 0, stream>>>(jbuf[jH2], WmT, jbuf[jH],  0, MODE_ZK, c1v, cdv, gatev);
        // jac = jac @ jac_zkp1 (per-batch B^T operand)
        bgemm_wmma<<<gG, gB, 0, stream>>>(jbuf[Jc],  jbuf[jH], jbuf[jH2],
                                          (long)DD * DD, MODE_PLAIN, nullptr, nullptr, nullptr);
        Jc = jH2;
    }
    finalize<<<((size_t)BB * DD * DD) / 256, 256, 0, stream>>>(zbuf, ldet, jbuf[Jc], out);
}
